// MNIST_Model_2_levels_wo_Att_10565619549011
// MI455X (gfx1250) — matrix operand order repaired and checked by reading, not compiled, not run on hardware
//
#include <hip/hip_runtime.h>
#include <math.h>

typedef __attribute__((ext_vector_type(16))) _Float16 v16h;
typedef __attribute__((ext_vector_type(8)))  _Float16 v8h;
typedef __attribute__((ext_vector_type(8)))  float    v8f;

#define THREADS 256
#define N_INST  8192
#define NUM_BAGS 256
#define EMB     1600
#define UNITS   512

// ---- workspace layout ----
#define WFRAG_WS_OFF ((NUM_BAGS * EMB + NUM_BAGS) * 4)   // 1639424 B (16B aligned)
#define WFRAG_HALVES (18432 + 1024)

// ---- dynamic LDS layout (bytes), all 32B aligned ----
#define W2F_OFF  0         // 18432 f16: conv2 A frags, [kc*4+m][lane][h]
#define A1F_OFF  36864     // 1024  f16: conv1 A frags, [m][lane][h]
#define XS_OFF   38912     // 784   f16: 28x28 input
#define XCOL_OFF 40480     // 676*16 f16: conv1 im2col (taps 9..15 = 0)
#define C1_OFF   62112     // 676*32 f16: conv1 out [p][oc]
#define H1_OFF   105376    // 169*32 f16: pooled1 [pix][ic]
#define C2_OFF   116192    // 121*64 f16: conv2 out [p][oc]
#define LDS_BYTES 131680

__global__ void zero_ws(float* __restrict__ p, int n) {
    int i = blockIdx.x * blockDim.x + threadIdx.x;
    if (i < n) p[i] = 0.f;
}

// Pre-swizzle conv weights into WMMA fragment order (f16), once per dispatch.
// A-frag map used here (16-bit A 16x32, same k order as B, not the ISA table order): lane L row = L%16, half h -> K = h + 16*(L>=16).
// Columns k>=9 of the conv1 A matrix are zero (K padded 9->32).
__global__ void prep_weights(const float* __restrict__ c1w, const float* __restrict__ c2w,
                             _Float16* __restrict__ wf) {
    int i = blockIdx.x * blockDim.x + threadIdx.x;
    if (i < 18432) {                      // conv2: 9 K-chunks x 4 M-tiles
        int f = i >> 9, pos = i & 511, ln = pos >> 4, h = pos & 15;
        int kc = f >> 2, m = f & 3;
        int k = kc * 32 + h + 16 * (ln >= 16);
        wf[i] = (_Float16)c2w[k * 64 + m * 16 + (ln & 15)];
    } else if (i < WFRAG_HALVES) {        // conv1: 1 K-chunk (9 of 32) x 2 M-tiles
        int j = i - 18432;
        int m = j >> 9, pos = j & 511, ln = pos >> 4, h = pos & 15;
        int k = h + 16 * (ln >= 16);
        float v = (k < 9) ? c1w[k * 32 + m * 16 + (ln & 15)] : 0.f;
        wf[i] = (_Float16)v;
    }
}

static __device__ __forceinline__ v8f wmma_f16(v16h a, v16h b, v8f c) {
    return __builtin_amdgcn_wmma_f32_16x16x32_f16(false, a, false, b, (short)0, c, false, false);
}

__launch_bounds__(THREADS)
__global__ void fused_conv_bag(const float* __restrict__ x,
                               const int*   __restrict__ first_lab,
                               const float* __restrict__ c1b,
                               const float* __restrict__ c2b,
                               const _Float16* __restrict__ wfrag,
                               float* __restrict__ sums, float* __restrict__ counts)
{
    extern __shared__ char smem[];
    _Float16* w2f  = (_Float16*)(smem + W2F_OFF);
    _Float16* a1fs = (_Float16*)(smem + A1F_OFF);
    _Float16* xs   = (_Float16*)(smem + XS_OFF);
    _Float16* xcol = (_Float16*)(smem + XCOL_OFF);
    _Float16* c1s  = (_Float16*)(smem + C1_OFF);
    _Float16* h1s  = (_Float16*)(smem + H1_OFF);
    _Float16* c2s  = (_Float16*)(smem + C2_OFF);

    const int tid  = threadIdx.x;
    const int lane = tid & 31;
    const int wv   = tid >> 5;           // 8 waves
    const int img  = blockIdx.x;
    const int arow = lane & 15;
    const int hiA  = (lane >= 16) ? 1 : 0;
    const int mrow = 8 * hiA;            // C fragment: element r <-> M = r + 8*hiA

    // Per-lane bias accumulator inits (C of D = A*B + C): bias depends on M only.
    v8f bia0, bia1;                      // conv1 oc tiles 0,1
    v8f cb0, cb1, cb2, cb3;              // conv2 oc tiles 0..3
    #pragma unroll
    for (int r = 0; r < 8; ++r) {
        bia0[r] = c1b[r + mrow];
        bia1[r] = c1b[16 + r + mrow];
        cb0[r]  = c2b[r + mrow];
        cb1[r]  = c2b[16 + r + mrow];
        cb2[r]  = c2b[32 + r + mrow];
        cb3[r]  = c2b[48 + r + mrow];
    }

    // ---------------- stage: weight frags (vector copy) + input ----------------
    {
        const uint4* wsrc = (const uint4*)wfrag;
        uint4*       wdst = (uint4*)smem;                  // W2F + A1F contiguous, 38912 B
        for (int i = tid; i < (WFRAG_HALVES * 2) / 16; i += THREADS) wdst[i] = wsrc[i];
    }
    const float* xin = x + img * 784;
    for (int i = tid; i < 784; i += THREADS) xs[i] = (_Float16)xin[i];
    __syncthreads();

    // ---------------- build conv1 im2col rows (16 halves per pixel, taps>=9 zero) -------
    for (int p = tid; p < 676; p += THREADS) {
        int oi = p / 26, oj = p % 26;
        v16h v = {};
        #pragma unroll
        for (int j = 0; j < 9; ++j) v[j] = xs[(oi + j / 3) * 28 + (oj + j % 3)];
        *(v16h*)(xcol + p * 16) = v;
    }
    __syncthreads();

    // ---------------- conv1 WMMA GEMM: D[oc,p] = W1[oc,k] * X[k,p] + b1  (K=9 pad 32) --
    const v16h a0  = *(const v16h*)(a1fs + lane * 16);
    const v16h a1f = *(const v16h*)(a1fs + 512 + lane * 16);

    for (int n = wv; n < 43; n += 8) {                   // 43 N-tiles cover 676 pixels
        int p  = n * 16 + arow;
        int pc = p > 675 ? 675 : p;
        // B rows 16..31 (lanes 16-31) multiply zero A columns -> no masking needed.
        v16h bv = *(const v16h*)(xcol + pc * 16);
        v8f d0 = wmma_f16(a0,  bv, bia0);
        v8f d1 = wmma_f16(a1f, bv, bia1);
        if (p <= 675) {
            #pragma unroll
            for (int r = 0; r < 8; ++r) {
                float v0 = d0[r]; v0 = v0 > 0.f ? v0 : 0.f;
                float v1 = d1[r]; v1 = v1 > 0.f ? v1 : 0.f;
                c1s[p * 32 + r + mrow]      = (_Float16)v0;
                c1s[p * 32 + 16 + r + mrow] = (_Float16)v1;
            }
        }
    }
    __syncthreads();

    // ---------------- maxpool 2x2: 26x26x32 -> 13x13x32 (v8h packed max) ---------------
    for (int i = tid; i < 13 * 13 * 4; i += THREADS) {   // groups of 8 channels
        int icg = i & 3; int pj = (i >> 2) % 13; int pi = i / 52;
        int r0 = 2 * pi, q0 = 2 * pj;
        const int co = icg * 8;
        v8h m  = *(const v8h*)(c1s + (r0 * 26 + q0) * 32 + co);
        v8h t1 = *(const v8h*)(c1s + (r0 * 26 + q0 + 1) * 32 + co);
        v8h t2 = *(const v8h*)(c1s + ((r0 + 1) * 26 + q0) * 32 + co);
        v8h t3 = *(const v8h*)(c1s + ((r0 + 1) * 26 + q0 + 1) * 32 + co);
        #pragma unroll
        for (int e = 0; e < 8; ++e) {
            _Float16 a = m[e];
            if (t1[e] > a) a = t1[e];
            if (t2[e] > a) a = t2[e];
            if (t3[e] > a) a = t3[e];
            m[e] = a;
        }
        *(v8h*)(h1s + (pi * 13 + pj) * 32 + co) = m;
    }
    __syncthreads();

    // ---------------- conv2 WMMA GEMM: 64 x 288 x 121 (9 K-chunks), bias in C ----------
    {
        int p  = wv * 16 + arow;                          // wave owns one 16-pixel N-tile
        int pc = p > 120 ? 120 : p;
        int oi = pc / 11, oj = pc % 11;
        v8f acc0 = cb0, acc1 = cb1, acc2 = cb2, acc3 = cb3;
        for (int kc = 0; kc < 9; ++kc) {
            // B frag: lane's 16 halves = one input pixel, 16 consecutive channels
            const _Float16* bp = h1s + ((oi + kc / 3) * 13 + (oj + kc % 3)) * 32 + 16 * hiA;
            v16h bv = *(const v16h*)bp;
            const _Float16* af = w2f + ((kc * 4) << 9) + (lane << 4);
            v16h a;
            a = *(const v16h*)(af);          acc0 = wmma_f16(a, bv, acc0);
            a = *(const v16h*)(af + 512);    acc1 = wmma_f16(a, bv, acc1);
            a = *(const v16h*)(af + 1024);   acc2 = wmma_f16(a, bv, acc2);
            a = *(const v16h*)(af + 1536);   acc3 = wmma_f16(a, bv, acc3);
        }
        if (p <= 120) {
            #pragma unroll
            for (int r = 0; r < 8; ++r) {
                float v;
                v = acc0[r]; v = v > 0.f ? v : 0.f; c2s[p * 64 + r + mrow]      = (_Float16)v;
                v = acc1[r]; v = v > 0.f ? v : 0.f; c2s[p * 64 + 16 + r + mrow] = (_Float16)v;
                v = acc2[r]; v = v > 0.f ? v : 0.f; c2s[p * 64 + 32 + r + mrow] = (_Float16)v;
                v = acc3[r]; v = v > 0.f ? v : 0.f; c2s[p * 64 + 48 + r + mrow] = (_Float16)v;
            }
        }
    }
    __syncthreads();

    // ---------------- pool2 (11x11->5x5) + per-bag segment sum -------------------------
    const int bag = first_lab[img];
    float* bagsum = sums + (size_t)bag * EMB;
    for (int i = tid; i < 5 * 5 * 8; i += THREADS) {     // groups of 8 channels
        int chg = i & 7; int pj = (i >> 3) % 5; int pi = i / 40;
        int r0 = 2 * pi, q0 = 2 * pj;
        const int co = chg * 8;
        v8h m  = *(const v8h*)(c2s + (r0 * 11 + q0) * 64 + co);
        v8h t1 = *(const v8h*)(c2s + (r0 * 11 + q0 + 1) * 64 + co);
        v8h t2 = *(const v8h*)(c2s + ((r0 + 1) * 11 + q0) * 64 + co);
        v8h t3 = *(const v8h*)(c2s + ((r0 + 1) * 11 + q0 + 1) * 64 + co);
        int base = pi * 320 + pj * 64 + co;              // NHWC flatten
        #pragma unroll
        for (int e = 0; e < 8; ++e) {
            _Float16 a = m[e];
            if (t1[e] > a) a = t1[e];
            if (t2[e] > a) a = t2[e];
            if (t3[e] > a) a = t3[e];
            unsafeAtomicAdd(&bagsum[base + e], (float)a);
        }
    }
    if (tid == 0) unsafeAtomicAdd(&counts[bag], 1.0f);
}

__launch_bounds__(512)
__global__ void finalize_head(const float* __restrict__ sums, const float* __restrict__ counts,
                              const float* __restrict__ w1, const float* __restrict__ b1,
                              const float* __restrict__ w2, const float* __restrict__ b2,
                              float* __restrict__ out)
{
    __shared__ float outer[EMB];
    __shared__ float red[512];
    int tid = threadIdx.x;
    for (int i = tid; i < EMB; i += 512) {
        float mx = -INFINITY;
        for (int bg = 0; bg < NUM_BAGS; ++bg) {
            float c = counts[bg];
            if (c > 0.f) {
                float v = sums[(size_t)bg * EMB + i] / c;
                if (v > mx) mx = v;
            }
        }
        outer[i] = mx;
    }
    __syncthreads();
    float acc = b1[tid];
    for (int e = 0; e < EMB; ++e) acc += outer[e] * w1[e * UNITS + tid];
    red[tid] = acc * w2[tid];
    __syncthreads();
    for (int s = 256; s > 0; s >>= 1) {
        if (tid < s) red[tid] += red[tid + s];
        __syncthreads();
    }
    if (tid == 0) out[0] = 1.f / (1.f + __expf(-(red[0] + b2[0])));
}

extern "C" void kernel_launch(void* const* d_in, const int* in_sizes, int n_in,
                              void* d_out, int out_size, void* d_ws, size_t ws_size,
                              hipStream_t stream) {
    const float* x   = (const float*)d_in[0];
    const int*   lab = (const int*)  d_in[1];
    const float* c1w = (const float*)d_in[2];
    const float* c1b = (const float*)d_in[3];
    const float* c2w = (const float*)d_in[4];
    const float* c2b = (const float*)d_in[5];
    const float* w1  = (const float*)d_in[6];
    const float* b1  = (const float*)d_in[7];
    const float* w2  = (const float*)d_in[8];
    const float* b2  = (const float*)d_in[9];

    float*    sums   = (float*)d_ws;                              // [256][1600]
    float*    counts = sums + NUM_BAGS * EMB;                     // [256]
    _Float16* wfrag  = (_Float16*)((char*)d_ws + WFRAG_WS_OFF);   // fragment-ordered weights
    int ztotal = NUM_BAGS * EMB + NUM_BAGS;

    zero_ws<<<(ztotal + 255) / 256, 256, 0, stream>>>(sums, ztotal);
    prep_weights<<<(WFRAG_HALVES + 255) / 256, 256, 0, stream>>>(c1w, c2w, wfrag);
    fused_conv_bag<<<N_INST, THREADS, LDS_BYTES, stream>>>(x, lab, c1b, c2b, wfrag,
                                                           sums, counts);
    finalize_head<<<1, 512, 0, stream>>>(sums, counts, w1, b1, w2, b2, (float*)d_out);
}
